// MessagePassingGraph_59124519797099
// MI455X (gfx1250) — matrix operand order repaired and checked by reading, not compiled, not run on hardware
//
#include <hip/hip_runtime.h>
#include <cstdint>
#include <cfloat>

// ---------------------------------------------------------------------------
// Problem constants (match reference)
// ---------------------------------------------------------------------------
#define NNODES 32000
#define NEDGES 320000
#define NG     8
#define DD     128

typedef __attribute__((ext_vector_type(16))) _Float16 v16h;
typedef __attribute__((ext_vector_type(8)))  _Float16 v8h;
typedef __attribute__((ext_vector_type(8)))  float    v8f;

// ---------------------------------------------------------------------------
// WMMA fragment helpers (CDNA5 v_wmma_f32_16x16x32_f16, wave32)
//
// A (16x32 f16): lane = hh*16 + r  (r = row, hh selects K-halves)
//   half i = A[r][k0 + hh*16 + i]
// B (32x16 f16): lane = hb*16 + c  (c = col)
//   half i = B[k0 + hb*16 + i][c]   -> pre-packed contiguous per lane
// C/D (16x16 f32, 8 VGPRs): elem i -> row = 8*(lane>>4) + i, col = lane&15
// ---------------------------------------------------------------------------
__device__ __forceinline__ v16h lda_frag(const _Float16* base, int stride,
                                         int m0, int k0) {
  const int lane = threadIdx.x & 31;
  const int r = lane & 15, hh = lane >> 4;
  const _Float16* p = base + (size_t)(m0 + r) * stride + k0 + hh * 16;
  v8h lo = *(const v8h*)p;
  v8h hi = *(const v8h*)(p + 8);
  return __builtin_shufflevector(lo, hi, 0, 1, 2, 3, 4, 5, 6, 7,
                                 8, 9, 10, 11, 12, 13, 14, 15);
}

__device__ __forceinline__ v16h ldb_frag(const _Float16* wp, int NT, int kt,
                                         int nt) {
  const int lane = threadIdx.x & 31;
  const _Float16* p = wp + ((((size_t)(kt * NT + nt)) * 32 + lane) << 4);
  return *(const v16h*)p;  // 32B contiguous per lane
}

__device__ __forceinline__ v8f wmma16(v16h a, v16h b, v8f c) {
  return __builtin_amdgcn_wmma_f32_16x16x32_f16(false, a, false, b, (short)0,
                                                c, false, false);
}

__device__ __forceinline__ void atomicMaxF32(float* addr, float val) {
  if (val >= 0.0f)
    atomicMax((int*)addr, __float_as_int(val));
  else
    atomicMin((unsigned int*)addr, (unsigned int)__float_as_int(val));
}

// ---------------------------------------------------------------------------
// Fused edge kernel: tile of 32 edges, 256 threads (8 waves)
//   X = concat(edge, nodes[send], nodes[recv], glob)  [32 x 512] f16 in LDS
//   H = relu(X @ W1 + b1)                             [32 x 256]
//   Enew = H @ W2 + b2   -> f32 out  + f16 into X[:,0:128]
//   Ha = relu(X' @ Wa1 + ba1)                         [32 x 128]
//   logit = Ha . wa2 + ba2                            [32]
// ---------------------------------------------------------------------------
__global__ __launch_bounds__(256) void edge_fused_kernel(
    const _Float16* __restrict__ edges16, const _Float16* __restrict__ nodes16,
    const _Float16* __restrict__ glob16, const int* __restrict__ senders,
    const int* __restrict__ receivers, const _Float16* __restrict__ w1p,
    const float* __restrict__ b1, const _Float16* __restrict__ w2p,
    const float* __restrict__ b2, const _Float16* __restrict__ wa1p,
    const float* __restrict__ ba1, const float* __restrict__ wa2,
    const float* __restrict__ ba2, float* __restrict__ enew,
    float* __restrict__ logits) {
  __shared__ _Float16 Xs[32 * 512];
  __shared__ _Float16 Hs[32 * 256];
  __shared__ _Float16 As[32 * 128];
  __shared__ float red[32][8];

  const int t = threadIdx.x;
  const int e0 = blockIdx.x * 32;

  // ---- gather: 8 threads per row, each copies 64 halves ----
  {
    const int r = t >> 3, q = t & 7, s = q >> 1, h = (q & 1) * 64;
    const int e = e0 + r;
    const _Float16* src;
    if (s == 0)
      src = edges16 + (size_t)e * DD;
    else if (s == 1)
      src = nodes16 + (size_t)senders[e] * DD;
    else if (s == 2)
      src = nodes16 + (size_t)receivers[e] * DD;
    else
      src = glob16 + (size_t)(e / (NEDGES / NG)) * DD;
    src += h;
    _Float16* dst = Xs + r * 512 + s * 128 + h;
#pragma unroll
    for (int j = 0; j < 64; j += 8) *(v8h*)(dst + j) = *(const v8h*)(src + j);
  }
  __syncthreads();

  const int wave = t >> 5;
  const int lane = t & 31;
  const int col16 = lane & 15;
  const int row8 = (lane >> 4) << 3;

  // ---- GEMM1: X[32x512] @ W1[512x256] -> relu -> Hs ----
  {
    const int rt = wave & 1;
    const int cb = (wave >> 1) * 4;  // 16 col tiles over 4 wave-groups
    v8f acc[4] = {};
    for (int kt = 0; kt < 16; ++kt) {
      v16h a = lda_frag(Xs, 512, rt * 16, kt * 32);
#pragma unroll
      for (int c = 0; c < 4; ++c) {
        v16h b = ldb_frag(w1p, 16, kt, cb + c);
        acc[c] = wmma16(a, b, acc[c]);
      }
    }
#pragma unroll
    for (int c = 0; c < 4; ++c) {
      const int n = (cb + c) * 16 + col16;
      const float bv = b1[n];
      const int rb = rt * 16 + row8;
#pragma unroll
      for (int i = 0; i < 8; ++i)
        Hs[(rb + i) * 256 + n] = (_Float16)fmaxf(acc[c][i] + bv, 0.0f);
    }
  }
  __syncthreads();

  // ---- GEMM2: H[32x256] @ W2[256x128] + b2 -> enew f32 + X[:,0:128] f16 ----
  {
    const int rt = wave & 1;
    const int cb = (wave >> 1) * 2;  // 8 col tiles
    v8f acc[2] = {};
    for (int kt = 0; kt < 8; ++kt) {
      v16h a = lda_frag(Hs, 256, rt * 16, kt * 32);
#pragma unroll
      for (int c = 0; c < 2; ++c) {
        v16h b = ldb_frag(w2p, 8, kt, cb + c);
        acc[c] = wmma16(a, b, acc[c]);
      }
    }
#pragma unroll
    for (int c = 0; c < 2; ++c) {
      const int n = (cb + c) * 16 + col16;
      const float bv = b2[n];
      const int rb = rt * 16 + row8;
#pragma unroll
      for (int i = 0; i < 8; ++i) {
        const float v = acc[c][i] + bv;
        enew[(size_t)(e0 + rb + i) * DD + n] = v;
        Xs[(rb + i) * 512 + n] = (_Float16)v;
      }
    }
  }
  __syncthreads();

  // ---- GEMM3: X'[32x512] @ Wa1[512x128] -> relu -> As ----
  {
    const int rt = wave & 1;
    const int cb = (wave >> 1) * 2;
    v8f acc[2] = {};
    for (int kt = 0; kt < 16; ++kt) {
      v16h a = lda_frag(Xs, 512, rt * 16, kt * 32);
#pragma unroll
      for (int c = 0; c < 2; ++c) {
        v16h b = ldb_frag(wa1p, 8, kt, cb + c);
        acc[c] = wmma16(a, b, acc[c]);
      }
    }
#pragma unroll
    for (int c = 0; c < 2; ++c) {
      const int n = (cb + c) * 16 + col16;
      const float bv = ba1[n];
      const int rb = rt * 16 + row8;
#pragma unroll
      for (int i = 0; i < 8; ++i)
        As[(rb + i) * 128 + n] = (_Float16)fmaxf(acc[c][i] + bv, 0.0f);
    }
  }
  __syncthreads();

  // ---- logit: per-row dot with wa2[128] ----
  {
    const int r = t >> 3, q = t & 7;
    float p = 0.0f;
#pragma unroll
    for (int j = 0; j < 16; ++j) {
      const int k = q * 16 + j;
      p += (float)As[r * 128 + k] * wa2[k];
    }
    red[r][q] = p;
  }
  __syncthreads();
  if (t < 32) {
    float s = ba2[0];
#pragma unroll
    for (int q = 0; q < 8; ++q) s += red[t][q];
    logits[e0 + t] = s;
  }
}

// ---------------------------------------------------------------------------
// Fused node kernel: tile of 32 nodes
//   X = concat(node, agg_sent/cnt, agg_recv/cnt, glob)  [32 x 512]
//   out = (relu(X@W1+b1))@W2 + b2  -> f32 out + f16 in-place update
// ---------------------------------------------------------------------------
__global__ __launch_bounds__(256) void node_fused_kernel(
    const _Float16* nodes16_in, const float* __restrict__ aggs,
    const float* __restrict__ aggr, const float* __restrict__ cnt_s,
    const float* __restrict__ cnt_r, const _Float16* __restrict__ glob16,
    const _Float16* __restrict__ w1p, const float* __restrict__ b1,
    const _Float16* __restrict__ w2p, const float* __restrict__ b2,
    float* __restrict__ nodes_out32, _Float16* nodes_out16) {
  __shared__ _Float16 Xs[32 * 512];
  __shared__ _Float16 Hs[32 * 256];

  const int t = threadIdx.x;
  const int n0 = blockIdx.x * 32;

  {
    const int r = t >> 3, q = t & 7, s = q >> 1, h = (q & 1) * 64;
    const int n = n0 + r;
    _Float16* dst = Xs + r * 512 + s * 128 + h;
    if (s == 0) {
      const _Float16* src = nodes16_in + (size_t)n * DD + h;
#pragma unroll
      for (int j = 0; j < 64; j += 8)
        *(v8h*)(dst + j) = *(const v8h*)(src + j);
    } else if (s == 3) {
      const _Float16* src = glob16 + (size_t)(n / (NNODES / NG)) * DD + h;
#pragma unroll
      for (int j = 0; j < 64; j += 8)
        *(v8h*)(dst + j) = *(const v8h*)(src + j);
    } else {
      const float* src = (s == 1 ? aggs : aggr) + (size_t)n * DD + h;
      const float inv = 1.0f / fmaxf((s == 1 ? cnt_s : cnt_r)[n], 1.0f);
      for (int j = 0; j < 64; j += 4) {
        float4 v = *(const float4*)(src + j);
        dst[j + 0] = (_Float16)(v.x * inv);
        dst[j + 1] = (_Float16)(v.y * inv);
        dst[j + 2] = (_Float16)(v.z * inv);
        dst[j + 3] = (_Float16)(v.w * inv);
      }
    }
  }
  __syncthreads();

  const int wave = t >> 5;
  const int lane = t & 31;
  const int col16 = lane & 15;
  const int row8 = (lane >> 4) << 3;

  {
    const int rt = wave & 1;
    const int cb = (wave >> 1) * 4;
    v8f acc[4] = {};
    for (int kt = 0; kt < 16; ++kt) {
      v16h a = lda_frag(Xs, 512, rt * 16, kt * 32);
#pragma unroll
      for (int c = 0; c < 4; ++c) {
        v16h b = ldb_frag(w1p, 16, kt, cb + c);
        acc[c] = wmma16(a, b, acc[c]);
      }
    }
#pragma unroll
    for (int c = 0; c < 4; ++c) {
      const int n = (cb + c) * 16 + col16;
      const float bv = b1[n];
      const int rb = rt * 16 + row8;
#pragma unroll
      for (int i = 0; i < 8; ++i)
        Hs[(rb + i) * 256 + n] = (_Float16)fmaxf(acc[c][i] + bv, 0.0f);
    }
  }
  __syncthreads();

  {
    const int rt = wave & 1;
    const int cb = (wave >> 1) * 2;
    v8f acc[2] = {};
    for (int kt = 0; kt < 8; ++kt) {
      v16h a = lda_frag(Hs, 256, rt * 16, kt * 32);
#pragma unroll
      for (int c = 0; c < 2; ++c) {
        v16h b = ldb_frag(w2p, 8, kt, cb + c);
        acc[c] = wmma16(a, b, acc[c]);
      }
    }
#pragma unroll
    for (int c = 0; c < 2; ++c) {
      const int n = (cb + c) * 16 + col16;
      const float bv = b2[n];
      const int rb = rt * 16 + row8;
#pragma unroll
      for (int i = 0; i < 8; ++i) {
        const float v = acc[c][i] + bv;
        nodes_out32[(size_t)(n0 + rb + i) * DD + n] = v;
        nodes_out16[(size_t)(n0 + rb + i) * DD + n] = (_Float16)v;
      }
    }
  }
}

// ---------------------------------------------------------------------------
// n2g / e2g: per-row MLP (128->256->128) + per-graph atomic accumulation
// ---------------------------------------------------------------------------
__global__ __launch_bounds__(256) void seg_mlp_sum_kernel(
    const _Float16* __restrict__ x16, int rows_per_graph,
    const _Float16* __restrict__ w1p, const float* __restrict__ b1,
    const _Float16* __restrict__ w2p, const float* __restrict__ b2,
    float* __restrict__ gsum) {
  __shared__ _Float16 Xs[32 * 128];
  __shared__ _Float16 Hs[32 * 256];

  const int t = threadIdx.x;
  const int r0 = blockIdx.x * 32;
  {
    const int r = t >> 3, h = (t & 7) * 16;
    const _Float16* src = x16 + (size_t)(r0 + r) * DD + h;
    _Float16* dst = Xs + r * 128 + h;
    *(v8h*)dst = *(const v8h*)src;
    *(v8h*)(dst + 8) = *(const v8h*)(src + 8);
  }
  __syncthreads();

  const int wave = t >> 5;
  const int lane = t & 31;
  const int col16 = lane & 15;
  const int row8 = (lane >> 4) << 3;

  {
    const int rt = wave & 1;
    const int cb = (wave >> 1) * 4;
    v8f acc[4] = {};
    for (int kt = 0; kt < 4; ++kt) {
      v16h a = lda_frag(Xs, 128, rt * 16, kt * 32);
#pragma unroll
      for (int c = 0; c < 4; ++c) {
        v16h b = ldb_frag(w1p, 16, kt, cb + c);
        acc[c] = wmma16(a, b, acc[c]);
      }
    }
#pragma unroll
    for (int c = 0; c < 4; ++c) {
      const int n = (cb + c) * 16 + col16;
      const float bv = b1[n];
      const int rb = rt * 16 + row8;
#pragma unroll
      for (int i = 0; i < 8; ++i)
        Hs[(rb + i) * 256 + n] = (_Float16)fmaxf(acc[c][i] + bv, 0.0f);
    }
  }
  __syncthreads();

  {
    const int rt = wave & 1;
    const int cb = (wave >> 1) * 2;
    v8f acc[2] = {};
    for (int kt = 0; kt < 8; ++kt) {
      v16h a = lda_frag(Hs, 256, rt * 16, kt * 32);
#pragma unroll
      for (int c = 0; c < 2; ++c) {
        v16h b = ldb_frag(w2p, 8, kt, cb + c);
        acc[c] = wmma16(a, b, acc[c]);
      }
    }
#pragma unroll
    for (int c = 0; c < 2; ++c) {
      const int n = (cb + c) * 16 + col16;
      const float bv = b2[n];
      const int rb = rt * 16 + row8;
#pragma unroll
      for (int i = 0; i < 8; ++i) {
        const int row = r0 + rb + i;
        atomicAdd(&gsum[(row / rows_per_graph) * DD + n], acc[c][i] + bv);
      }
    }
  }
}

// ---------------------------------------------------------------------------
// Global MLP (8 rows, f32 scalar — negligible FLOPs)
// ---------------------------------------------------------------------------
__global__ __launch_bounds__(256) void global_mlp_kernel(
    const float* __restrict__ n2gsum, const float* __restrict__ e2gsum,
    const float* __restrict__ glob_cur, float inv_n, float inv_e,
    const float* __restrict__ W1, const float* __restrict__ b1,
    const float* __restrict__ W2, const float* __restrict__ b2,
    float* __restrict__ glob_out32, _Float16* __restrict__ glob_out16) {
  __shared__ float X[NG][384];
  __shared__ float H[NG][256];
  const int t = threadIdx.x;
  for (int idx = t; idx < NG * 384; idx += 256) {
    const int g = idx / 384, k = idx % 384;
    float v;
    if (k < 128)
      v = n2gsum[g * DD + k] * inv_n;
    else if (k < 256)
      v = e2gsum[g * DD + (k - 128)] * inv_e;
    else
      v = glob_cur[g * DD + (k - 256)];
    X[g][k] = v;
  }
  __syncthreads();
  for (int idx = t; idx < NG * 256; idx += 256) {
    const int g = idx / 256, j = idx % 256;
    float a = b1[j];
    for (int k = 0; k < 384; ++k) a += X[g][k] * W1[k * 256 + j];
    H[g][j] = fmaxf(a, 0.0f);
  }
  __syncthreads();
  for (int idx = t; idx < NG * DD; idx += 256) {
    const int g = idx / DD, n = idx % DD;
    float a = b2[n];
    for (int j = 0; j < 256; ++j) a += H[g][j] * W2[j * DD + n];
    glob_out32[g * DD + n] = a;
    glob_out16[g * DD + n] = (_Float16)a;
  }
}

// ---------------------------------------------------------------------------
// Small utility kernels
// ---------------------------------------------------------------------------
__global__ void fill_f32(float* p, float v, int n) {
  const int i = blockIdx.x * 256 + threadIdx.x;
  if (i < n) p[i] = v;
}

__global__ void cvt16_kernel(const float* __restrict__ s,
                             _Float16* __restrict__ d, int n) {
  const int i = blockIdx.x * 256 + threadIdx.x;
  if (i < n) d[i] = (_Float16)s[i];
}

// Pack W[KxN] f32 into B-fragment order: half index
//   (((kt*NT + nt)*32 + lane)*16 + i), lane = ((k%32)/16)*16 + n%16, i = k%16
__global__ void pack_w_kernel(const float* __restrict__ W,
                              _Float16* __restrict__ out, int K, int N) {
  const int i = blockIdx.x * 256 + threadIdx.x;
  if (i >= K * N) return;
  const int k = i / N, n = i % N;
  const int kt = k >> 5, rk = k & 31, hb = rk >> 4, ii = rk & 15;
  const int nt = n >> 4, c = n & 15;
  const int NT = N >> 4;
  out[(((size_t)(kt * NT + nt)) * 32 + hb * 16 + c) * 16 + ii] =
      (_Float16)W[i];
}

__global__ void count_kernel(const int* __restrict__ senders,
                             const int* __restrict__ receivers, float* cs,
                             float* cr) {
  const int i = blockIdx.x * 256 + threadIdx.x;
  if (i < NEDGES) {
    atomicAdd(&cs[senders[i]], 1.0f);
    atomicAdd(&cr[receivers[i]], 1.0f);
  }
}

__global__ void seg_max_kernel(const float* __restrict__ logits,
                               const int* __restrict__ recv, float* segmax) {
  const int i = blockIdx.x * 256 + threadIdx.x;
  if (i < NEDGES) atomicMaxF32(&segmax[recv[i]], logits[i]);
}

__global__ void seg_exp_kernel(float* logits, const int* __restrict__ recv,
                               const float* __restrict__ segmax,
                               float* segden) {
  const int i = blockIdx.x * 256 + threadIdx.x;
  if (i < NEDGES) {
    const float e = expf(logits[i] - segmax[recv[i]]);
    logits[i] = e;
    atomicAdd(&segden[recv[i]], e);
  }
}

// edges *= softmax weight (in place, f32) ; mirror to f16 ; scatter-add to
// per-node sums for segment means. One thread = 4 floats of one edge row.
__global__ void scale_agg_kernel(float* edges32, _Float16* edges16,
                                 const float* __restrict__ logits,
                                 const float* __restrict__ segden,
                                 const int* __restrict__ senders,
                                 const int* __restrict__ receivers,
                                 float* aggs, float* aggr) {
  const int gid = blockIdx.x * 256 + threadIdx.x;  // NEDGES*32 total
  const int e = gid >> 5;
  const int off = (gid & 31) * 4;
  const int rc = receivers[e];
  const int sd = senders[e];
  const float w = logits[e] / segden[rc];
  float4 v = *(const float4*)(edges32 + (size_t)e * DD + off);
  v.x *= w; v.y *= w; v.z *= w; v.w *= w;
  *(float4*)(edges32 + (size_t)e * DD + off) = v;
  _Float16* d16 = edges16 + (size_t)e * DD + off;
  d16[0] = (_Float16)v.x; d16[1] = (_Float16)v.y;
  d16[2] = (_Float16)v.z; d16[3] = (_Float16)v.w;
  float* as = aggs + (size_t)sd * DD + off;
  float* ar = aggr + (size_t)rc * DD + off;
  atomicAdd(as + 0, v.x); atomicAdd(as + 1, v.y);
  atomicAdd(as + 2, v.z); atomicAdd(as + 3, v.w);
  atomicAdd(ar + 0, v.x); atomicAdd(ar + 1, v.y);
  atomicAdd(ar + 2, v.z); atomicAdd(ar + 3, v.w);
}

// ---------------------------------------------------------------------------
// Host orchestration
// ---------------------------------------------------------------------------
extern "C" void kernel_launch(void* const* d_in, const int* in_sizes, int n_in,
                              void* d_out, int out_size, void* d_ws,
                              size_t ws_size, hipStream_t stream) {
  (void)in_sizes; (void)n_in; (void)out_size; (void)ws_size;

  const float* in_nodes = (const float*)d_in[0];
  const float* in_edges = (const float*)d_in[1];
  const float* in_glob  = (const float*)d_in[2];
  const int* senders    = (const int*)d_in[3];
  const int* receivers  = (const int*)d_in[4];
  // params flatten order (sorted dict keys): attn, e2g, edge, glob, n2g, node;
  // each MLP flattens as W1,b1,W2,b2.  24 arrays per layer.
  auto P = [&](int l, int idx) -> const float* {
    return (const float*)d_in[5 + l * 24 + idx];
  };

  // ---- workspace carve (all 256B aligned) ----
  char* w = (char*)d_ws;
  auto carve = [&](size_t bytes) {
    char* p = w;
    w += (bytes + 255) & ~(size_t)255;
    return p;
  };
  const size_t PACK_PER_LAYER = 524288;  // halves
  _Float16* wpack   = (_Float16*)carve(2 * PACK_PER_LAYER * sizeof(_Float16));
  _Float16* nodes16 = (_Float16*)carve((size_t)NNODES * DD * 2);
  _Float16* edges16 = (_Float16*)carve((size_t)NEDGES * DD * 2);
  _Float16* glob16  = (_Float16*)carve((size_t)NG * DD * 2);
  float* logits = (float*)carve((size_t)NEDGES * 4);
  float* segmax = (float*)carve((size_t)NNODES * 4);
  float* segden = (float*)carve((size_t)NNODES * 4);
  float* cnt_s  = (float*)carve((size_t)NNODES * 4);
  float* cnt_r  = (float*)carve((size_t)NNODES * 4);
  float* aggs   = (float*)carve((size_t)NNODES * DD * 4);
  float* aggr   = (float*)carve((size_t)NNODES * DD * 4);
  float* n2gsum = (float*)carve((size_t)NG * DD * 4);
  float* e2gsum = (float*)carve((size_t)NG * DD * 4);

  // packed-weight offsets within a layer (halves)
  const size_t OFF_ATTN_W1 = 0;       // 512x128
  const size_t OFF_E2G_W1  = 65536;   // 128x256
  const size_t OFF_E2G_W2  = 98304;   // 256x128
  const size_t OFF_EDGE_W1 = 131072;  // 512x256
  const size_t OFF_EDGE_W2 = 262144;  // 256x128
  const size_t OFF_N2G_W1  = 294912;  // 128x256
  const size_t OFF_N2G_W2  = 327680;  // 256x128
  const size_t OFF_NODE_W1 = 360448;  // 512x256
  const size_t OFF_NODE_W2 = 491520;  // 256x128

  float* out_nodes = (float*)d_out;
  float* out_edges = out_nodes + (size_t)NNODES * DD;
  float* out_glob  = out_edges + (size_t)NEDGES * DD;

  // ---- one-time prep ----
  cvt16_kernel<<<(NNODES * DD + 255) / 256, 256, 0, stream>>>(
      in_nodes, nodes16, NNODES * DD);
  cvt16_kernel<<<(NEDGES * DD + 255) / 256, 256, 0, stream>>>(
      in_edges, edges16, NEDGES * DD);
  cvt16_kernel<<<(NG * DD + 255) / 256, 256, 0, stream>>>(in_glob, glob16,
                                                          NG * DD);
  fill_f32<<<(NNODES + 255) / 256, 256, 0, stream>>>(cnt_s, 0.0f, NNODES);
  fill_f32<<<(NNODES + 255) / 256, 256, 0, stream>>>(cnt_r, 0.0f, NNODES);
  count_kernel<<<(NEDGES + 255) / 256, 256, 0, stream>>>(senders, receivers,
                                                         cnt_s, cnt_r);
  struct PW { int idx; int K; int N; size_t off; };
  const PW tab[9] = {
      {0, 512, 128, OFF_ATTN_W1}, {4, 128, 256, OFF_E2G_W1},
      {6, 256, 128, OFF_E2G_W2},  {8, 512, 256, OFF_EDGE_W1},
      {10, 256, 128, OFF_EDGE_W2}, {16, 128, 256, OFF_N2G_W1},
      {18, 256, 128, OFF_N2G_W2}, {20, 512, 256, OFF_NODE_W1},
      {22, 256, 128, OFF_NODE_W2}};
  for (int l = 0; l < 2; ++l)
    for (int i = 0; i < 9; ++i)
      pack_w_kernel<<<(tab[i].K * tab[i].N + 255) / 256, 256, 0, stream>>>(
          P(l, tab[i].idx), wpack + l * PACK_PER_LAYER + tab[i].off, tab[i].K,
          tab[i].N);

  // ---- two message-passing layers ----
  for (int l = 0; l < 2; ++l) {
    _Float16* wp = wpack + l * PACK_PER_LAYER;
    fill_f32<<<(NNODES + 255) / 256, 256, 0, stream>>>(segmax, -FLT_MAX,
                                                       NNODES);
    fill_f32<<<(NNODES + 255) / 256, 256, 0, stream>>>(segden, 0.0f, NNODES);
    fill_f32<<<(NNODES * DD + 255) / 256, 256, 0, stream>>>(aggs, 0.0f,
                                                            NNODES * DD);
    fill_f32<<<(NNODES * DD + 255) / 256, 256, 0, stream>>>(aggr, 0.0f,
                                                            NNODES * DD);
    fill_f32<<<(NG * DD + 255) / 256, 256, 0, stream>>>(n2gsum, 0.0f, NG * DD);
    fill_f32<<<(NG * DD + 255) / 256, 256, 0, stream>>>(e2gsum, 0.0f, NG * DD);

    // edge + attention MLPs (fused, WMMA)
    edge_fused_kernel<<<NEDGES / 32, 256, 0, stream>>>(
        edges16, nodes16, glob16, senders, receivers, wp + OFF_EDGE_W1,
        P(l, 9), wp + OFF_EDGE_W2, P(l, 11), wp + OFF_ATTN_W1, P(l, 1),
        P(l, 2), P(l, 3), out_edges, logits);

    // segment softmax over receivers
    seg_max_kernel<<<NEDGES / 256, 256, 0, stream>>>(logits, receivers,
                                                     segmax);
    seg_exp_kernel<<<NEDGES / 256, 256, 0, stream>>>(logits, receivers,
                                                     segmax, segden);
    scale_agg_kernel<<<NEDGES * 32 / 256, 256, 0, stream>>>(
        out_edges, edges16, logits, segden, senders, receivers, aggs, aggr);

    // node MLP (fused, WMMA); updates nodes16 in place
    node_fused_kernel<<<NNODES / 32, 256, 0, stream>>>(
        nodes16, aggs, aggr, cnt_s, cnt_r, glob16, wp + OFF_NODE_W1, P(l, 21),
        wp + OFF_NODE_W2, P(l, 23), out_nodes, nodes16);

    // node->global and edge->global MLPs + per-graph sums
    seg_mlp_sum_kernel<<<NNODES / 32, 256, 0, stream>>>(
        nodes16, NNODES / NG, wp + OFF_N2G_W1, P(l, 17), wp + OFF_N2G_W2,
        P(l, 19), n2gsum);
    seg_mlp_sum_kernel<<<NEDGES / 32, 256, 0, stream>>>(
        edges16, NEDGES / NG, wp + OFF_E2G_W1, P(l, 5), wp + OFF_E2G_W2,
        P(l, 7), e2gsum);

    // global MLP
    global_mlp_kernel<<<1, 256, 0, stream>>>(
        n2gsum, e2gsum, (l == 0) ? in_glob : out_glob,
        1.0f / (float)(NNODES / NG), 1.0f / (float)(NEDGES / NG), P(l, 12),
        P(l, 13), P(l, 14), P(l, 15), out_glob, glob16);
  }
}
